// DotProductAttention_31679678775913
// MI455X (gfx1250) — matrix operand order repaired and checked by reading, not compiled, not run on hardware
//
#include <hip/hip_runtime.h>
#include <math.h>

typedef __attribute__((ext_vector_type(16))) _Float16 v16h;
typedef __attribute__((ext_vector_type(8)))  float    v8f;
typedef __attribute__((ext_vector_type(4)))  float    f4;

#define B_      8
#define L_      2048
#define N_      64
#define TQ      16
#define NTHREADS 256
#define NWAVES  8

// One block: (batch, 16-query tile). Waves split the 2048 key columns into
// 32-wide chunks. Scores via v_wmma_f32_16x16x32_f16 (K=64 in two steps),
// softmax over full rows staged in LDS (320KB WGP LDS makes single-pass easy),
// attended via WMMA again, weights streamed out as non-temporal b128 stores
// (the 134 MB store stream is the roofline floor: ~5.9 us at 23.3 TB/s).
// All Linear(1,64) constants are pre-converted to f16 and held in registers so
// the inner tile loops carry no LDS dependencies.
__global__ __launch_bounds__(NTHREADS)
void attn_aug_kernel(const float* __restrict__ query,
                     const float* __restrict__ key,
                     const float* __restrict__ value,
                     const float* __restrict__ W,      // [N,1] flat
                     const float* __restrict__ bias,   // [N]
                     float* __restrict__ attended,     // [B,L,N]
                     float* __restrict__ weights)      // [B,L,L]
{
    extern __shared__ float smem[];
    float* sS   = smem;                 // TQ*L   exp/scores stripe (128 KB)
    float* sAtt = sS   + TQ * L_;       // TQ*N   attended accumulators
    float* sW   = sAtt + TQ * N_;       // N   (one-time A-frag build only)
    float* sB   = sW   + N_;            // N
    float* sRed = sB   + N_;            // NTHREADS reduction scratch
    float* sMax = sRed + NTHREADS;      // TQ
    float* sInv = sMax + TQ;            // TQ

    const int tid  = threadIdx.x;
    const int lane = tid & 31;
    const int wid  = tid >> 5;
    const int qt   = blockIdx.x;        // 0..127
    const int bat  = blockIdx.y;        // 0..7
    const int hi   = (lane & 16) ? 1 : 0;
    const int M    = lane & 15;

    const int qbase = bat * L_ + qt * TQ;
    const int kbase = bat * L_;

    for (int i = tid; i < TQ * N_; i += NTHREADS) sAtt[i] = 0.0f;
    if (tid < N_) { sW[tid] = W[tid]; sB[tid] = bias[tid]; }

    // ---- Loop-invariant B-fragment constants, pre-converted to f16, in
    // registers. B 32x16: lane col = lane&15; this code's contiguous K order: element h -> K = h+16*hi.
    // So this lane only ever needs n = hi*16 + h (and +32): two contiguous
    // 16-float chunks of W and bias -> b128 loads, converted once.
    v16h wb0, bb0, wb1, bb1;
    {
        const f4* wp0 = (const f4*)(W    + hi * 16);
        const f4* bp0 = (const f4*)(bias + hi * 16);
        const f4* wp1 = (const f4*)(W    + 32 + hi * 16);
        const f4* bp1 = (const f4*)(bias + 32 + hi * 16);
        #pragma unroll
        for (int q4 = 0; q4 < 4; ++q4) {
            f4 w0 = wp0[q4], b0 = bp0[q4], w1 = wp1[q4], b1 = bp1[q4];
            #pragma unroll
            for (int e = 0; e < 4; ++e) {
                wb0[q4 * 4 + e] = (_Float16)w0[e];
                bb0[q4 * 4 + e] = (_Float16)b0[e];
                wb1[q4 * 4 + e] = (_Float16)w1[e];
                bb1[q4 * 4 + e] = (_Float16)b1[e];
            }
        }
    }
    __syncthreads();

    // ---- Augmented Q tile as f16 A-fragments (K = n, split 0..31 / 32..63).
    // A 16x32: lane M = lane&15; this code's contiguous K order (same as B): element h -> K = h+16*hi
    const float xq = query[qbase + M];
    v16h a0, a1;
    #pragma unroll
    for (int h = 0; h < 16; ++h) {
        int n = h + hi * 16;
        a0[h] = (_Float16)(xq * sW[n]      + sB[n]);
        a1[h] = (_Float16)(xq * sW[n + 32] + sB[n + 32]);
    }
    // Phase-3 V-fragment constants (n = g*16 + lane&15), f16 scalars.
    _Float16 wg_h[4], bg_h[4];
    #pragma unroll
    for (int g = 0; g < 4; ++g) {
        wg_h[g] = (_Float16)sW[g * 16 + (lane & 15)];
        bg_h[g] = (_Float16)sB[g * 16 + (lane & 15)];
    }

    // ---- Phase 1: S[16 x 2048] = Qaug * Kaug^T into LDS.
    for (int ct = wid; ct < L_ / 32; ct += NWAVES) {
        #pragma unroll
        for (int t = 0; t < 2; ++t) {
            const int jb = ct * 32 + t * 16;
            const _Float16 yh = (_Float16)key[kbase + jb + (lane & 15)];
            v16h kb0, kb1;
            #pragma unroll
            for (int h = 0; h < 16; ++h) {
                kb0[h] = yh * wb0[h] + bb0[h];   // v_pk_fma_f16
                kb1[h] = yh * wb1[h] + bb1[h];
            }
            v8f c = {};
            c = __builtin_amdgcn_wmma_f32_16x16x32_f16(false, a0, false, kb0,
                                                       (short)0, c, false, false);
            c = __builtin_amdgcn_wmma_f32_16x16x32_f16(false, a1, false, kb1,
                                                       (short)0, c, false, false);
            // C/D layout: VGPR v -> row M = v + 8*hi, col = lane&15
            #pragma unroll
            for (int v = 0; v < 8; ++v)
                sS[(v + hi * 8) * L_ + jb + (lane & 15)] = c[v];
        }
    }
    __syncthreads();

    // ---- Phase 2: softmax, vectorized f4 (ds_load_b128/ds_store_b128).
    {
        const int r = tid >> 4, li = tid & 15;
        f4* rowv = (f4*)(sS + r * L_);
        float mx = -1e30f;
        for (int c = li; c < L_ / 4; c += 16) {
            f4 vv = rowv[c];
            mx = fmaxf(mx, fmaxf(fmaxf(vv[0], vv[1]), fmaxf(vv[2], vv[3])));
        }
        sRed[tid] = mx;
        __syncthreads();
        if (tid < TQ) {
            float m = -1e30f;
            #pragma unroll
            for (int k = 0; k < 16; ++k) m = fmaxf(m, sRed[tid * 16 + k]);
            sMax[tid] = m;
        }
        __syncthreads();
        const float rm = sMax[r];
        float sm = 0.0f;
        for (int c = li; c < L_ / 4; c += 16) {
            f4 vv = rowv[c];
            f4 e;
            e[0] = __expf(vv[0] - rm);
            e[1] = __expf(vv[1] - rm);
            e[2] = __expf(vv[2] - rm);
            e[3] = __expf(vv[3] - rm);
            rowv[c] = e;            // keep unnormalized exp in LDS
            sm += (e[0] + e[1]) + (e[2] + e[3]);
        }
        sRed[tid] = sm;
        __syncthreads();
        if (tid < TQ) {
            float z = 0.0f;
            #pragma unroll
            for (int k = 0; k < 16; ++k) z += sRed[tid * 16 + k];
            sInv[tid] = 1.0f / z;
        }
        __syncthreads();
    }

    // ---- Phase 3: attended = P * Vaug via WMMA (normalize at store).
    {
        v8f acc[4] = {};
        for (int ct = wid; ct < L_ / 32; ct += NWAVES) {
            v16h pA;
            #pragma unroll
            for (int h = 0; h < 16; ++h) {
                int jl = h + hi * 16;
                pA[h] = (_Float16)sS[M * L_ + ct * 32 + jl];
            }
            // 16 consecutive value floats per lane -> 4x global_load_b128,
            // converted to f16 once and reused for all 4 column groups.
            v16h tvh;
            const f4* vsrc = (const f4*)(value + kbase + ct * 32 + hi * 16);
            #pragma unroll
            for (int q4 = 0; q4 < 4; ++q4) {
                f4 t = vsrc[q4];
                #pragma unroll
                for (int e = 0; e < 4; ++e) tvh[q4 * 4 + e] = (_Float16)t[e];
            }
            #pragma unroll
            for (int g = 0; g < 4; ++g) {
                v16h vB;
                #pragma unroll
                for (int h = 0; h < 16; ++h)
                    vB[h] = tvh[h] * wg_h[g] + bg_h[g];   // v_pk_fma_f16
                acc[g] = __builtin_amdgcn_wmma_f32_16x16x32_f16(false, pA, false, vB,
                                                                (short)0, acc[g],
                                                                false, false);
            }
        }
        #pragma unroll
        for (int g = 0; g < 4; ++g) {
            #pragma unroll
            for (int v = 0; v < 8; ++v)
                atomicAdd(&sAtt[(v + hi * 8) * N_ + g * 16 + (lane & 15)], acc[g][v]);
        }
    }
    __syncthreads();

    // ---- Phase 4: stores, b128-wide.
    // attended tile is contiguous: rows qbase..qbase+15 x 64 floats = 1024 floats.
    {
        // 256 threads x float4 == 1024 floats, one shot.
        const int m = tid >> 4;                 // (tid*4)/64
        f4 vv = ((const f4*)sAtt)[tid] * sInv[m];
        ((f4*)(attended + (size_t)qbase * N_))[tid] = vv;
    }
    // weights: 128 KB per block, streaming non-temporal b128 stores.
    for (int m = 0; m < TQ; ++m) {
        const float inv = sInv[m];
        const f4* src = (const f4*)(sS + m * L_);
        f4* dst = (f4*)(weights + (size_t)(qbase + m) * L_);
        for (int c = tid; c < L_ / 4; c += NTHREADS) {
            f4 vv = src[c] * inv;
            __builtin_nontemporal_store(vv, dst + c);
        }
    }
}

extern "C" void kernel_launch(void* const* d_in, const int* in_sizes, int n_in,
                              void* d_out, int out_size, void* d_ws, size_t ws_size,
                              hipStream_t stream) {
    const float* query = (const float*)d_in[0];
    const float* key_  = (const float*)d_in[1];
    const float* value = (const float*)d_in[2];
    const float* W     = (const float*)d_in[3];
    const float* bias  = (const float*)d_in[4];

    float* out      = (float*)d_out;
    float* attended = out;                           // [B,L,N] first
    float* weights  = out + (size_t)B_ * L_ * N_;    // then [B,L,L]

    const size_t shmem =
        (size_t)(TQ * L_ + TQ * N_ + N_ + N_ + NTHREADS + TQ + TQ) * sizeof(float);

    // >64KB dynamic LDS: raise the per-kernel cap (host-side, deterministic,
    // safe under graph capture).
    hipFuncSetAttribute((const void*)attn_aug_kernel,
                        hipFuncAttributeMaxDynamicSharedMemorySize, (int)shmem);

    attn_aug_kernel<<<dim3(L_ / TQ, B_), NTHREADS, shmem, stream>>>(
        query, key_, value, W, bias, attended, weights);
}
